// transformerBlock_37245956391467
// MI455X (gfx1250) — matrix operand order repaired and checked by reading, not compiled, not run on hardware
//
#include <hip/hip_runtime.h>
#include <math.h>

typedef unsigned short u16;
typedef unsigned int u32;
typedef __attribute__((ext_vector_type(16))) __bf16 v16bf;
typedef __attribute__((ext_vector_type(8)))  float  v8f;

static __device__ __forceinline__ u16 f2bf(float f) {
    union { float f; u32 u; } c; c.f = f;
    u32 u = c.u;
    u += 0x7fffu + ((u >> 16) & 1u);      // round-to-nearest-even
    return (u16)(u >> 16);
}

static __device__ __forceinline__ v8f vzero8() {
    v8f z = {0.f, 0.f, 0.f, 0.f, 0.f, 0.f, 0.f, 0.f};
    return z;
}

static __device__ __forceinline__ v8f wmma_bf16(v16bf a, v16bf b, v8f c) {
    return __builtin_amdgcn_wmma_f32_16x16x32_bf16(
        /*neg_a=*/false, a, /*neg_b=*/false, b,
        /*c_mod=*/(short)0, c, /*reuse_a=*/false, /*reuse_b=*/false);
}

// Async copy 16B global -> LDS, per lane. ASYNCcnt-tracked (CDNA5).
static __device__ __forceinline__ void async_b128(u16* lds_dst, const u16* gsrc) {
    unsigned la = (unsigned)(uintptr_t)lds_dst;   // low 32 bits = LDS offset
    asm volatile("global_load_async_to_lds_b128 %0, %1, off"
                 :: "v"(la), "v"(gsrc) : "memory");
}
static __device__ __forceinline__ void wait_async0() {
    asm volatile("s_wait_asynccnt 0" ::: "memory");
}

// ---------------------------------------------------------------------------
// Weight convert + transpose: W[K][N] fp32  ->  WT[N][K] bf16
// ---------------------------------------------------------------------------
__global__ __launch_bounds__(256) void wt_kernel(const float* __restrict__ W,
                                                 u16* __restrict__ WT,
                                                 int K, int N) {
    size_t idx = (size_t)blockIdx.x * 256 + threadIdx.x;
    size_t total = (size_t)K * N;
    if (idx < total) {
        int k = (int)(idx / N);
        int n = (int)(idx % N);
        WT[(size_t)n * K + k] = f2bf(W[idx]);
    }
}

// ---------------------------------------------------------------------------
// LayerNorm over D=768, one block (256 thr) per row, bf16 output
// ---------------------------------------------------------------------------
__global__ __launch_bounds__(256) void ln_kernel(const float* __restrict__ in,
                                                 const float* __restrict__ g,
                                                 const float* __restrict__ bta,
                                                 u16* __restrict__ out) {
    __shared__ float red[256];
    const int row = blockIdx.x;
    const int tid = threadIdx.x;
    const float* p = in + (size_t)row * 768;
    float x0 = p[tid], x1 = p[tid + 256], x2 = p[tid + 512];
    red[tid] = x0 + x1 + x2;
    __syncthreads();
    for (int o = 128; o > 0; o >>= 1) { if (tid < o) red[tid] += red[tid + o]; __syncthreads(); }
    float mu = red[0] * (1.f / 768.f);
    __syncthreads();
    float d0 = x0 - mu, d1 = x1 - mu, d2 = x2 - mu;
    red[tid] = d0 * d0 + d1 * d1 + d2 * d2;
    __syncthreads();
    for (int o = 128; o > 0; o >>= 1) { if (tid < o) red[tid] += red[tid + o]; __syncthreads(); }
    float rstd = rsqrtf(red[0] * (1.f / 768.f) + 1e-3f);
    u16* q = out + (size_t)row * 768;
    q[tid]       = f2bf(d0 * rstd * g[tid]       + bta[tid]);
    q[tid + 256] = f2bf(d1 * rstd * g[tid + 256] + bta[tid + 256]);
    q[tid + 512] = f2bf(d2 * rstd * g[tid + 512] + bta[tid + 512]);
}

// ---------------------------------------------------------------------------
// Tiled bf16 WMMA GEMM: C[M][N] = A[M][K] * BT[N][K]^T + bias, epilogue MODE
//   EPI_QK   : store bf16, headed layout [B,H,S,dh]
//   EPI_VT   : store bf16, transposed headed layout [B,H,dh,S]
//   EPI_GELU : gelu(acc+bias) -> bf16 [M][N]
//   EPI_RESID: acc+bias+aux   -> fp32 [M][N]
// Tile: BM=128, BN=128, BK=32; 8 waves, each 32x64 (2x4 wmma tiles).
// Double-buffered LDS filled with global_load_async_to_lds_b128: tile kt+1
// streams in (ASYNCcnt) while tile kt feeds the WMMAs.
// ---------------------------------------------------------------------------
constexpr int EPI_QK = 0, EPI_VT = 1, EPI_GELU = 2, EPI_RESID = 3;
constexpr int GBM = 128, GBN = 128, GBK = 32, GLD = 40; // padded k-stride (u16)

template <int MODE>
__global__ __launch_bounds__(256) void gemm_bf16(const u16* __restrict__ A,
                                                 const u16* __restrict__ BT,
                                                 const float* __restrict__ bias,
                                                 const float* __restrict__ aux,
                                                 void* __restrict__ out,
                                                 int M, int N, int K) {
    __shared__ __align__(16) u16 As[2][GBM * GLD];
    __shared__ __align__(16) u16 Bs[2][GBN * GLD];
    const int tid = threadIdx.x;
    const int lane = tid & 31;
    const int wid = tid >> 5;
    const int hf = (lane >> 4) & 1;
    const int l16 = lane & 15;
    const int m0 = blockIdx.y * GBM;
    const int n0 = blockIdx.x * GBN;
    const int wm = (wid & 3) * 32;
    const int wn = (wid >> 2) * 64;

    v8f acc[2][4];
#pragma unroll
    for (int i = 0; i < 2; i++)
#pragma unroll
        for (int j = 0; j < 4; j++) acc[i][j] = vzero8();

    const int nk = K / GBK;

    auto issue_tile = [&](int kt, int buf) {
        const int k0 = kt * GBK;
#pragma unroll
        for (int i = 0; i < 2; ++i) {
            int c = tid + i * 256;            // 512 16B chunks per tile
            int row = c >> 2, c8 = (c & 3) * 8;
            async_b128(&As[buf][row * GLD + c8], &A[(size_t)(m0 + row) * K + k0 + c8]);
            async_b128(&Bs[buf][row * GLD + c8], &BT[(size_t)(n0 + row) * K + k0 + c8]);
        }
    };

    issue_tile(0, 0);
    wait_async0();
    __syncthreads();

    for (int kt = 0; kt < nk; ++kt) {
        const int buf = kt & 1;
        if (kt + 1 < nk) issue_tile(kt + 1, buf ^ 1);   // overlap with compute

        v16bf afr[2], bfr[4];
        const int klo = hf * 8;
#pragma unroll
        for (int i = 0; i < 2; i++) {
            const u16* p = &As[buf][(wm + i * 16 + l16) * GLD];
            uint4* d = (uint4*)&afr[i];
            d[0] = *(const uint4*)&p[klo];
            d[1] = *(const uint4*)&p[16 + klo];
        }
        const int kb = hf * 8;
#pragma unroll
        for (int j = 0; j < 4; j++) {
            const u16* p = &Bs[buf][(wn + j * 16 + l16) * GLD];
            uint4* d = (uint4*)&bfr[j];
            d[0] = *(const uint4*)&p[kb];
            d[1] = *(const uint4*)&p[kb + 16];
        }
#pragma unroll
        for (int i = 0; i < 2; i++)
#pragma unroll
            for (int j = 0; j < 4; j++)
                acc[i][j] = wmma_bf16(afr[i], bfr[j], acc[i][j]);

        wait_async0();      // next tile landed (had full compute to hide)
        __syncthreads();
    }

    // epilogue
#pragma unroll
    for (int i = 0; i < 2; i++)
#pragma unroll
        for (int j = 0; j < 4; j++)
#pragma unroll
            for (int r = 0; r < 8; r++) {
                int mrow = m0 + wm + i * 16 + r + 8 * hf;
                int ncol = n0 + wn + j * 16 + l16;
                float v = acc[i][j][r] + bias[ncol];
                if (MODE == EPI_QK || MODE == EPI_VT) {
                    int b = mrow >> 10, s = mrow & 1023;  // S = 1024
                    int h = ncol / 192, d = ncol % 192;   // dh = 192
                    size_t o;
                    if (MODE == EPI_QK)
                        o = ((size_t)(b * 4 + h) * 1024 + s) * 192 + d;
                    else
                        o = ((size_t)(b * 4 + h) * 192 + d) * 1024 + s;
                    ((u16*)out)[o] = f2bf(v);
                } else if (MODE == EPI_GELU) {
                    float t = 0.7978845608f * (v + 0.044715f * v * v * v);
                    float gl = 0.5f * v * (1.f + tanhf(t));
                    ((u16*)out)[(size_t)mrow * N + ncol] = f2bf(gl);
                } else {
                    size_t o = (size_t)mrow * N + ncol;
                    ((float*)out)[o] = v + aux[o];
                }
            }
}

// ---------------------------------------------------------------------------
// Flash attention, bf16 WMMA.  Q,K: [B,H,S,dh] bf16; VT: [B,H,dh,S] bf16.
// One WG = 128 queries (8 waves x 16 rows). Streams 32-key tiles through
// DOUBLE-BUFFERED LDS via async-to-LDS (tile j+1 fills while tile j is
// computed). Writes resid = x + heads (fused residual); each (b,h) block
// owns a disjoint dh-column range of D.
// ---------------------------------------------------------------------------
constexpr int PLD = 32;  // P staging k-stride (u16), 16B-aligned rows

__global__ __launch_bounds__(256) void attn_kernel(const u16* __restrict__ Q,
                                                   const u16* __restrict__ Kh,
                                                   const u16* __restrict__ VT,
                                                   const float* __restrict__ X,
                                                   float* __restrict__ R) {
    constexpr int S = 1024, DH = 192, D = 768, H = 4;
    __shared__ __align__(16) u16 Ks[2][32 * 200];     // 32 keys x 192 (pad 200)
    __shared__ __align__(16) u16 Vs[2][192 * 40];     // 192 dh-rows x 32 keys (pad 40)
    __shared__ __align__(16) u16 Ps[8 * 16 * PLD];    // per-wave P staging 16x32

    const int tid = threadIdx.x, lane = tid & 31, wid = tid >> 5;
    const int hf = (lane >> 4) & 1, l16 = lane & 15;
    const int b = blockIdx.z, h = blockIdx.y;
    const int bh = b * H + h;
    const int q0 = blockIdx.x * 128;
    const u16* qp = Q  + (size_t)bh * S * DH;
    const u16* kp = Kh + (size_t)bh * S * DH;
    const u16* vp = VT + (size_t)bh * DH * S;

    // Q fragments for this wave's 16 rows (kept in registers for all 32 tiles)
    v16bf aq[6];
    const int qrow = q0 + wid * 16 + l16;
    const int klo = hf * 8;
#pragma unroll
    for (int kk = 0; kk < 6; kk++) {
        const u16* p = qp + (size_t)qrow * DH + kk * 32;
        uint4* d = (uint4*)&aq[kk];
        d[0] = *(const uint4*)&p[klo];
        d[1] = *(const uint4*)&p[16 + klo];
    }

    // async K/V tile fill: K 32x192, VT 192x32 (768 16B chunks each)
    auto issue_kv = [&](int j, int buf) {
#pragma unroll
        for (int i = 0; i < 3; ++i) {
            int c = tid + i * 256;
            int kr = c / 24, kc = (c % 24) * 8;
            async_b128(&Ks[buf][kr * 200 + kc], &kp[(size_t)(j * 32 + kr) * DH + kc]);
            int vr = c >> 2, vc = (c & 3) * 8;
            async_b128(&Vs[buf][vr * 40 + vc], &vp[(size_t)vr * S + j * 32 + vc]);
        }
    };

    v8f o[12];
    float m[8], l[8];
#pragma unroll
    for (int t = 0; t < 12; t++) o[t] = vzero8();
#pragma unroll
    for (int r = 0; r < 8; r++) { m[r] = -3.0e38f; l[r] = 0.f; }

    issue_kv(0, 0);
    wait_async0();
    __syncthreads();

    for (int j = 0; j < S / 32; ++j) {
        const int buf = j & 1;
        if (j + 1 < S / 32) issue_kv(j + 1, buf ^ 1);   // overlap with compute

        // scores: 16x32 = q(16x192) @ K^T, 12 WMMAs
        v8f sc[2] = {vzero8(), vzero8()};
#pragma unroll
        for (int kk = 0; kk < 6; kk++)
#pragma unroll
            for (int nb = 0; nb < 2; nb++) {
                v16bf bk;
                const u16* p = &Ks[buf][(nb * 16 + l16) * 200 + kk * 32 + hf * 8];
                uint4* d = (uint4*)&bk;
                d[0] = *(const uint4*)&p[0];
                d[1] = *(const uint4*)&p[16];
                sc[nb] = wmma_bf16(aq[kk], bk, sc[nb]);
            }

        // online softmax per row (rows live in 16-lane halves)
        const float scale = 0.07216878364870323f;   // 1/sqrt(192)
        float corr[8];
#pragma unroll
        for (int r = 0; r < 8; r++) {
            float rm = -3.0e38f;
#pragma unroll
            for (int nb = 0; nb < 2; nb++) { sc[nb][r] *= scale; rm = fmaxf(rm, sc[nb][r]); }
#pragma unroll
            for (int msk = 8; msk >= 1; msk >>= 1) rm = fmaxf(rm, __shfl_xor(rm, msk, 32));
            float mn = fmaxf(m[r], rm);
            corr[r] = __expf(m[r] - mn);
            float rs = 0.f;
#pragma unroll
            for (int nb = 0; nb < 2; nb++) {
                sc[nb][r] = __expf(sc[nb][r] - mn);
                rs += sc[nb][r];
            }
#pragma unroll
            for (int msk = 8; msk >= 1; msk >>= 1) rs += __shfl_xor(rs, msk, 32);
            l[r] = l[r] * corr[r] + rs;
            m[r] = mn;
        }

        // stage P (C-layout -> row-major LDS) for A-fragment reload
        u16* pw = &Ps[wid * 16 * PLD];
#pragma unroll
        for (int nb = 0; nb < 2; nb++)
#pragma unroll
            for (int r = 0; r < 8; r++)
                pw[(r + 8 * hf) * PLD + nb * 16 + l16] = f2bf(sc[nb][r]);
        __syncthreads();

        // rescale running output
#pragma unroll
        for (int t = 0; t < 12; t++)
#pragma unroll
            for (int r = 0; r < 8; r++) o[t][r] *= corr[r];

        v16bf ap;
        {
            const u16* p = &pw[l16 * PLD];
            uint4* d = (uint4*)&ap;
            d[0] = *(const uint4*)&p[klo];
            d[1] = *(const uint4*)&p[16 + klo];
        }
        // O += P(16x32) @ V(32x192): 12 WMMAs
#pragma unroll
        for (int nv = 0; nv < 12; nv++) {
            v16bf bv;
            const u16* p = &Vs[buf][(nv * 16 + l16) * 40 + hf * 8];
            uint4* d = (uint4*)&bv;
            d[0] = *(const uint4*)&p[0];
            d[1] = *(const uint4*)&p[16];
            o[nv] = wmma_bf16(ap, bv, o[nv]);
        }

        wait_async0();      // next K/V tile landed (hidden behind compute)
        __syncthreads();
    }

    // finalize: resid = x + O / l
#pragma unroll
    for (int r = 0; r < 8; r++) l[r] = 1.f / l[r];
#pragma unroll
    for (int nv = 0; nv < 12; nv++)
#pragma unroll
        for (int r = 0; r < 8; r++) {
            int s = q0 + wid * 16 + r + 8 * hf;
            int d = h * DH + nv * 16 + l16;
            size_t idx = (size_t)(b * S + s) * D + d;
            R[idx] = X[idx] + o[nv][r] * l[r];
        }
}

// ---------------------------------------------------------------------------
extern "C" void kernel_launch(void* const* d_in, const int* in_sizes, int n_in,
                              void* d_out, int out_size, void* d_ws, size_t ws_size,
                              hipStream_t stream) {
    const float* x     = (const float*)d_in[0];
    const float* ln1_g = (const float*)d_in[1];
    const float* ln1_b = (const float*)d_in[2];
    const float* Wq    = (const float*)d_in[3];
    const float* bq    = (const float*)d_in[4];
    const float* Wk    = (const float*)d_in[5];
    const float* bk    = (const float*)d_in[6];
    const float* Wv    = (const float*)d_in[7];
    const float* bv    = (const float*)d_in[8];
    const float* ln2_g = (const float*)d_in[9];
    const float* ln2_b = (const float*)d_in[10];
    const float* W1    = (const float*)d_in[11];
    const float* b1    = (const float*)d_in[12];
    const float* W2    = (const float*)d_in[13];
    const float* b2    = (const float*)d_in[14];

    const int Bn = 8, S = 1024, D = 768, HD = 3072, M = Bn * S;

    char* ws = (char*)d_ws;
    size_t off = 0;
    auto alloc = [&](size_t bytes) -> void* {
        void* p = ws + off;
        off = (off + bytes + 255) & ~(size_t)255;
        return p;
    };
    u16*   yb    = (u16*)alloc((size_t)M * D * 2);
    u16*   WqT   = (u16*)alloc((size_t)D * D * 2);
    u16*   WkT   = (u16*)alloc((size_t)D * D * 2);
    u16*   WvT   = (u16*)alloc((size_t)D * D * 2);
    u16*   W1T   = (u16*)alloc((size_t)D * HD * 2);
    u16*   W2T   = (u16*)alloc((size_t)HD * D * 2);
    u16*   qh    = (u16*)alloc((size_t)M * D * 2);
    u16*   kh    = (u16*)alloc((size_t)M * D * 2);
    u16*   vh    = (u16*)alloc((size_t)M * D * 2);
    float* resid = (float*)alloc((size_t)M * D * 4);
    u16*   y2b   = (u16*)alloc((size_t)M * D * 2);
    u16*   h1    = (u16*)alloc((size_t)M * HD * 2);

    dim3 blk(256);
    wt_kernel<<<(D * D + 255) / 256, blk, 0, stream>>>(Wq, WqT, D, D);
    wt_kernel<<<(D * D + 255) / 256, blk, 0, stream>>>(Wk, WkT, D, D);
    wt_kernel<<<(D * D + 255) / 256, blk, 0, stream>>>(Wv, WvT, D, D);
    wt_kernel<<<(D * HD + 255) / 256, blk, 0, stream>>>(W1, W1T, D, HD);
    wt_kernel<<<(HD * D + 255) / 256, blk, 0, stream>>>(W2, W2T, HD, D);

    ln_kernel<<<M, blk, 0, stream>>>(x, ln1_g, ln1_b, yb);

    gemm_bf16<EPI_QK><<<dim3(D / GBN, M / GBM), blk, 0, stream>>>(yb, WqT, bq, nullptr, qh, M, D, D);
    gemm_bf16<EPI_QK><<<dim3(D / GBN, M / GBM), blk, 0, stream>>>(yb, WkT, bk, nullptr, kh, M, D, D);
    gemm_bf16<EPI_VT><<<dim3(D / GBN, M / GBM), blk, 0, stream>>>(yb, WvT, bv, nullptr, vh, M, D, D);

    attn_kernel<<<dim3(S / 128, 4, Bn), blk, 0, stream>>>(qh, kh, vh, x, resid);

    ln_kernel<<<M, blk, 0, stream>>>(resid, ln2_g, ln2_b, y2b);

    gemm_bf16<EPI_GELU><<<dim3(HD / GBN, M / GBM), blk, 0, stream>>>(y2b, W1T, b1, nullptr, h1, M, HD, D);
    gemm_bf16<EPI_RESID><<<dim3(D / GBN, M / GBM), blk, 0, stream>>>(h1, W2T, b2, resid, (float*)d_out, M, D, HD);
}
